// QuantLinear_54065048322158
// MI455X (gfx1250) — matrix operand order repaired and checked by reading, not compiled, not run on hardware
//
#include <hip/hip_runtime.h>
#include <hip/hip_bf16.h>
#include <stdint.h>

// GPTQ 4-bit dequant + GEMM for MI455X (gfx1250, wave32, WMMA).
// out[4096,11008] = x[4096,4096](f16) @ dequant(qweight)(f16) + bias
// Double-buffered: async A-tile DMA + packed-f16 dequant of group kg+1
// co-execute with the 32 WMMAs of group kg.

typedef __attribute__((ext_vector_type(16))) _Float16 v16h;
typedef __attribute__((ext_vector_type(8)))  _Float16 v8h;
typedef __attribute__((ext_vector_type(2)))  _Float16 v2h;
typedef __attribute__((ext_vector_type(8)))  float    v8f;

#define IN_F  4096
#define OUT_F 11008
#define NTOK  4096
#define GS    128
#define BM    128
#define BN    128
#define LDA   136     // padded row stride (halfs): 272B = 16B-aligned, breaks bank conflicts
#define LDB   136

// Stage one K-group: issue async global->LDS copy of the x tile and
// dequantize the 4-bit weight tile into LDS (N-major).
__device__ __forceinline__ void stage_tile(
    const _Float16* __restrict__ x,
    const int*      __restrict__ qweight,
    const int*      __restrict__ qzeros,
    const _Float16* __restrict__ scales,
    _Float16 (*__restrict__ As)[LDA],
    _Float16 (*__restrict__ Bt)[LDB],
    int m0, int n0, int kg, int tid)
{
  // ---- async DMA: x tile (128x128 f16 = 32KB), tracked by ASYNCcnt ----
  const _Float16* xg = x + (size_t)m0 * IN_F + (size_t)kg * GS;
#pragma unroll
  for (int t = 0; t < 8; ++t) {
    int idx = tid + t * 256;               // 2048 x 16B chunks
    int r   = idx >> 4;                    // row in tile
    int c8  = (idx & 15) << 3;             // col (halfs)
    unsigned lds  = (unsigned)(size_t)(&As[r][c8]);
    unsigned goff = (unsigned)((r * IN_F + c8) * 2);
    asm volatile("global_load_async_to_lds_b128 %0, %1, %2"
                 :: "v"(lds), "v"(goff), "s"(xg)
                 : "memory");
  }

  // ---- dequant: 16 packed rows x 128 cols int4 -> Bt[n][k] ----
  // f16 trick: (0x6400 | w) == 1024+w as f16; s*(w-z) = s*h - s*(1024+z).
  const int* qwt = qweight + (size_t)(kg * 16) * OUT_F + n0;
#pragma unroll
  for (int t = 0; t < 8; ++t) {
    int idx  = tid + t * 256;
    int n    = idx & 127;                  // col within tile
    int kp   = idx >> 7;                   // packed-K row 0..15
    int gcol = n0 + n;
    int q    = qwt[kp * OUT_F + n];
    int zw   = qzeros[kg * (OUT_F / 8) + (gcol >> 3)];
    int z    = ((zw >> ((gcol & 7) * 4)) & 15) + 1;      // GPTQ zero + 1
    _Float16 s  = scales[kg * OUT_F + gcol];
    _Float16 hz = (_Float16)((float)s * (float)(1024 + z));
    v2h sv  = {s, s};
    v2h hzv = {hz, hz};
    v8h w;
#pragma unroll
    for (int j = 0; j < 4; ++j) {
      unsigned tt = ((unsigned)q) >> (8 * j);
      unsigned packed = (tt & 0xFu) | ((tt << 12) & 0xF0000u) | 0x64006400u;
      v2h h = __builtin_bit_cast(v2h, packed);           // {1024+w0, 1024+w1}
      v2h r = h * sv - hzv;                              // v_pk_fma_f16
      w[2 * j]     = r[0];
      w[2 * j + 1] = r[1];
    }
    *(v8h*)(&Bt[n][kp << 3]) = w;          // 16B ds_store
  }
}

__launch_bounds__(256, 2)
__global__ void gptq_wmma_kernel(const _Float16* __restrict__ x,
                                 const int*      __restrict__ qweight,
                                 const int*      __restrict__ qzeros,
                                 const _Float16* __restrict__ scales,
                                 const _Float16* __restrict__ bias,
                                 float*          __restrict__ out) {
  __shared__ _Float16 As[2][BM][LDA];      // ping-pong x tiles
  __shared__ _Float16 Bt[2][BN][LDB];      // ping-pong dequantized W tiles

  const int tid   = threadIdx.x;
  const int lane  = tid & 31;
  const int wid   = tid >> 5;              // 0..7
  const int waveM = wid & 3;               // 4 waves along M
  const int waveN = wid >> 2;              // 2 waves along N
  const int lrow  = lane & 15;
  const int lhalf = lane >> 4;

  const int n0 = blockIdx.x * BN;
  const int m0 = blockIdx.y * BM;

  v8f acc[2][4];
#pragma unroll
  for (int mt = 0; mt < 2; ++mt)
#pragma unroll
    for (int nt = 0; nt < 4; ++nt)
      acc[mt][nt] = (v8f){0.f, 0.f, 0.f, 0.f, 0.f, 0.f, 0.f, 0.f};

  const int ngroups = IN_F / GS;           // 32 K-tiles

  // prologue: fill buffer 0
  stage_tile(x, qweight, qzeros, scales, As[0], Bt[0], m0, n0, 0, tid);
  asm volatile("s_wait_asynccnt 0" ::: "memory");
  __syncthreads();

  for (int kg = 0; kg < ngroups; ++kg) {
    const int cur = kg & 1;

    // stage next group into the other buffer; overlaps WMMAs below
    if (kg + 1 < ngroups)
      stage_tile(x, qweight, qzeros, scales, As[cur ^ 1], Bt[cur ^ 1],
                 m0, n0, kg + 1, tid);
    if (kg + 2 < ngroups)
      __builtin_prefetch(qweight + (size_t)((kg + 2) * 16) * OUT_F + n0 + (tid << 2), 0, 1);

    // ---- 4 WMMA K-steps of 32 over buffer `cur` ----
#pragma unroll
    for (int kk = 0; kk < 4; ++kk) {
      const int k0    = kk * 32;
      const int base0 = lhalf ? 8 : 0;     // A: lanes16-31 hold K=8..15,24..31
      const int kb    = lhalf * 8;         // B: lanes16-31 hold K=8..15,24..31

      v16h a[2], b[4];
#pragma unroll
      for (int mt = 0; mt < 2; ++mt) {
        int row = waveM * 32 + mt * 16 + lrow;
        v8h lo = *(const v8h*)(&As[cur][row][k0 + base0]);
        v8h hi = *(const v8h*)(&As[cur][row][k0 + base0 + 16]);
        a[mt] = __builtin_shufflevector(lo, hi,
                  0,1,2,3,4,5,6,7,8,9,10,11,12,13,14,15);
      }
#pragma unroll
      for (int nt = 0; nt < 4; ++nt) {
        int col = waveN * 64 + nt * 16 + lrow;
        v8h lo = *(const v8h*)(&Bt[cur][col][k0 + kb]);
        v8h hi = *(const v8h*)(&Bt[cur][col][k0 + kb + 16]);
        b[nt] = __builtin_shufflevector(lo, hi,
                  0,1,2,3,4,5,6,7,8,9,10,11,12,13,14,15);
      }
#pragma unroll
      for (int mt = 0; mt < 2; ++mt)
#pragma unroll
        for (int nt = 0; nt < 4; ++nt)
          acc[mt][nt] = __builtin_amdgcn_wmma_f32_16x16x32_f16(
              false, a[mt], false, b[nt],
              (short)0, acc[mt][nt], false, false);
    }

    // next buffer's DMA must land; barrier also orders dequant ds_stores
    // and protects buffer reuse two iterations out
    asm volatile("s_wait_asynccnt 0" ::: "memory");
    __syncthreads();
  }

  // ---- epilogue: bias add, f32 store per 16x16 C/D layout ----
#pragma unroll
  for (int mt = 0; mt < 2; ++mt) {
#pragma unroll
    for (int nt = 0; nt < 4; ++nt) {
      int col = n0 + waveN * 64 + nt * 16 + lrow;
      float bv = (float)bias[col];
      int rbase = m0 + waveM * 32 + mt * 16 + lhalf * 8;
#pragma unroll
      for (int v = 0; v < 8; ++v)
        out[(size_t)(rbase + v) * OUT_F + col] = acc[mt][nt][v] + bv;
    }
  }
}

extern "C" void kernel_launch(void* const* d_in, const int* in_sizes, int n_in,
                              void* d_out, int out_size, void* d_ws, size_t ws_size,
                              hipStream_t stream) {
  const _Float16* x       = (const _Float16*)d_in[0];
  const int*      qweight = (const int*)d_in[1];
  const int*      qzeros  = (const int*)d_in[2];
  const _Float16* scales  = (const _Float16*)d_in[3];
  // d_in[4] = g_idx (implicit: k / GROUPSIZE, so unused)
  const _Float16* bias    = (const _Float16*)d_in[5];
  float*          out     = (float*)d_out;

  dim3 grid(OUT_F / BN, NTOK / BM);        // (86, 32)
  gptq_wmma_kernel<<<grid, 256, 0, stream>>>(x, qweight, qzeros, scales, bias, out);
}
